// MixtureOfSoftmaxes_6975026889246
// MI455X (gfx1250) — matrix operand order repaired and checked by reading, not compiled, not run on hardware
//
#include <hip/hip_runtime.h>
#include <hip/hip_bf16.h>
#include <math.h>

typedef __attribute__((ext_vector_type(16))) __bf16 v16bf;
typedef __attribute__((ext_vector_type(8)))  __bf16 v8bf;
typedef __attribute__((ext_vector_type(8)))  float  v8f;
typedef __attribute__((ext_vector_type(4)))  unsigned int u32x4;
typedef __attribute__((ext_vector_type(4)))  int i32x4;
typedef __attribute__((ext_vector_type(8)))  int i32x8;

__device__ __forceinline__ unsigned short f2bf(float f) {
  unsigned int u = __float_as_uint(f);
  u += 0x7fffu + ((u >> 16) & 1u);          // round-to-nearest-even
  return (unsigned short)(u >> 16);
}

// A fragment (16x32 bf16): lane row m=lane%16; elements K = 8*khalf+(0..7), 16+8*khalf+(0..7)
__device__ __forceinline__ v16bf ldfragA(const unsigned short* row, int khalf) {
  v8bf lo = *(const v8bf*)(row + 8 * khalf);
  v8bf hi = *(const v8bf*)(row + 16 + 8 * khalf);
  return __builtin_shufflevector(lo, hi, 0,1,2,3,4,5,6,7,8,9,10,11,12,13,14,15);
}
// B fragment (32x16 bf16): lane col n=lane%16; elements K = 8*khalf+(0..7), 16+8*khalf+(0..7)
__device__ __forceinline__ v16bf ldfragB(const unsigned short* row, int khalf) {
  v8bf lo = *(const v8bf*)(row + 8 * khalf);
  v8bf hi = *(const v8bf*)(row + 16 + 8 * khalf);
  return __builtin_shufflevector(lo, hi, 0,1,2,3,4,5,6,7,8,9,10,11,12,13,14,15);
}

// ---- CDNA5 async global->LDS staging (ASYNCcnt path, ISA 08_async_tensor §4) ----
// Generic pointers to LDS carry the LDS byte offset in their low 32 bits.
__device__ __forceinline__ void async_ld_b32(void* lds, const void* g) {
  unsigned int l = (unsigned int)(unsigned long long)lds;
  unsigned long long ga = (unsigned long long)g;
  asm volatile("global_load_async_to_lds_b32 %0, %1, off" :: "v"(l), "v"(ga) : "memory");
}
__device__ __forceinline__ void wait_asynccnt0() {
  asm volatile("s_wait_asynccnt 0x0" ::: "memory");
}

// ---- CDNA5 Tensor Data Mover: 2D tile load Global->LDS (TENSORcnt path, ISA 08 §7-8) ----
// Loads tile_d1 rows x tile_d0 bf16 elements from a row-major tensor (row length
// tensor_d0, row stride stride_d0 elements) into LDS, inserting 16B of padding after
// every 64B row (pad_interval=3 -> 8x8B, pad_amount=3 -> 4 DWORDs) to match the
// sB[rows][40] layout (tile_d0 must be 32 bf16 = 64B).
__device__ __forceinline__ void tdm_load_2d_bf16(
    unsigned int lds_off, const void* gaddr,
    unsigned int tensor_d0, unsigned int tensor_d1,
    unsigned int tile_d0, unsigned int tile_d1, unsigned int stride_d0) {
  unsigned long long ga = (unsigned long long)gaddr;
  u32x4 g0;
  g0[0] = 1u;                                                       // count=1 (valid D#)
  g0[1] = lds_off;                                                  // lds_addr
  g0[2] = (unsigned int)ga;                                         // global_addr[31:0]
  g0[3] = (unsigned int)((ga >> 32) & 0x01FFFFFFu) | 0x80000000u;   // addr[56:32] | type=2
  i32x8 g1;
  g1[0] = (int)((1u << 16) | (1u << 20) | (3u << 22) | (3u << 25)); // data_size=2B, pad on
  g1[1] = (int)((tensor_d0 & 0xFFFFu) << 16);                       // tensor_dim0[15:0]
  g1[2] = (int)(((tensor_d0 >> 16) & 0xFFFFu) | ((tensor_d1 & 0xFFFFu) << 16));
  g1[3] = (int)(((tensor_d1 >> 16) & 0xFFFFu) | ((tile_d0 & 0xFFFFu) << 16));
  g1[4] = (int)(tile_d1 & 0xFFFFu);                                 // tile_dim1, tile_dim2=0
  g1[5] = (int)stride_d0;                                           // dim0_stride[31:0]
  g1[6] = 0;
  g1[7] = 0;
  i32x4 gz4 = {0, 0, 0, 0};                     // D# groups 2/3 unused (2D tensor)
  i32x8 gz8 = {0, 0, 0, 0, 0, 0, 0, 0};
  // clang-23 / therock toolchain: 6-arg form (g0, g1, g2, g3, extra, cpol)
  __builtin_amdgcn_tensor_load_to_lds(g0, g1, gz4, gz4, gz8, 0);
}

// ---------------- conversion kernels ----------------
__global__ void k_cvt_bf16(const float* __restrict__ in, unsigned short* __restrict__ out,
                           size_t n) {
  size_t i = (size_t)blockIdx.x * blockDim.x + threadIdx.x;
  if (i < n) out[i] = f2bf(in[i]);
}

// out[c][r] = bf16(in[r][c]) : in fp32 [R][C] -> out bf16 [C][R]
__global__ void k_transpose_cvt(const float* __restrict__ in, unsigned short* __restrict__ out,
                                int R, int C) {
  size_t i = (size_t)blockIdx.x * blockDim.x + threadIdx.x;
  if (i >= (size_t)R * C) return;
  size_t c = i / (size_t)R, r = i % (size_t)R;
  out[i] = f2bf(in[r * (size_t)C + c]);
}

// ---------------- prior: sigmoid(hidden @ prior_w + b), renormalized ----------------
__global__ __launch_bounds__(256) void k_prior(const float* __restrict__ hidden,
                                               const float* __restrict__ pw,
                                               const float* __restrict__ pb,
                                               float* __restrict__ prior) {
  const int H = 1024;
  const int lane = threadIdx.x & 31, wave = threadIdx.x >> 5;
  const size_t tok = (size_t)blockIdx.x * 8 + wave;
  float a[8] = {0.f, 0.f, 0.f, 0.f, 0.f, 0.f, 0.f, 0.f};
  for (int k = lane; k < H; k += 32) {
    float hv = hidden[tok * H + k];
#pragma unroll
    for (int n = 0; n < 8; ++n) a[n] += hv * pw[k * 8 + n];
  }
#pragma unroll
  for (int n = 0; n < 8; ++n) {
#pragma unroll
    for (int off = 16; off >= 1; off >>= 1) a[n] += __shfl_xor(a[n], off, 32);
  }
  float tot = 0.f;
#pragma unroll
  for (int n = 0; n < 8; ++n) {
    a[n] = 1.f / (1.f + __expf(-(a[n] + pb[n])));
    tot += a[n];
  }
  float inv = 1.f / (tot + 1e-8f);
  if (lane == 0) {
#pragma unroll
    for (int n = 0; n < 8; ++n) prior[tok * 8 + n] = a[n] * inv;
  }
}

// ---------------- GEMM1: latent = tanh(hidden @ latent_w + b), bf16 out ----------------
// A: hidden bf16 [8192][1024]; Bt: latent_w^T bf16 [8192][1024]; C: [8192][8192] bf16
// block tile 64(M) x 256(N), 8 waves (4 m x 2 n), wave tile 16x128 (8 WMMA tiles), K-step 32
__global__ __launch_bounds__(256) void k_latent_gemm_tanh(
    const unsigned short* __restrict__ A, const unsigned short* __restrict__ Bt,
    const float* __restrict__ lbias, unsigned short* __restrict__ C) {
  const int K = 1024, N = 8192;
  __shared__ unsigned short sA[64][40];    // padded rows (80B) to spread LDS banks
  __shared__ unsigned short sB[256][40];
  const int tid = threadIdx.x;
  const int lane = tid & 31, wave = tid >> 5;
  const int wm = wave >> 1, wn = wave & 1;
  const int khalf = lane >> 4;
  const int bm = blockIdx.y * 64;
  const int bn = blockIdx.x * 256;
  v8f acc[8] = {};
  for (int k0 = 0; k0 < K; k0 += 32) {
    __syncthreads();
    {  // stage A: 64x32, 8 elems (16B) per thread
      int e = tid * 8, m = e >> 5, kk = e & 31;
      *(uint4*)&sA[m][kk] = *(const uint4*)&A[(size_t)(bm + m) * K + k0 + kk];
    }
    {  // stage B: 256x32, one full row (64B) per thread
      int n = tid;
      const uint4* g = (const uint4*)&Bt[(size_t)(bn + n) * K + k0];
      uint4* s = (uint4*)&sB[n][0];
      s[0] = g[0]; s[1] = g[1]; s[2] = g[2]; s[3] = g[3];
    }
    __syncthreads();
    // load all fragments first, then issue the WMMA chain (one dscnt wait, full overlap)
    v16bf af = ldfragA(&sA[wm * 16 + (lane & 15)][0], khalf);
    v16bf bfm[8];
#pragma unroll
    for (int t = 0; t < 8; ++t)
      bfm[t] = ldfragB(&sB[wn * 128 + t * 16 + (lane & 15)][0], khalf);
#pragma unroll
    for (int t = 0; t < 8; ++t)
      acc[t] = __builtin_amdgcn_wmma_f32_16x16x32_bf16(false, af, false, bfm[t],
                                                       (short)0, acc[t], false, false);
  }
#pragma unroll
  for (int t = 0; t < 8; ++t) {
    int col = bn + wn * 128 + t * 16 + (lane & 15);
    float bv = lbias[col];
#pragma unroll
    for (int v = 0; v < 8; ++v) {
      int row = bm + wm * 16 + v + khalf * 8;  // C layout: row = v + 8*(lane>=16)
      C[(size_t)row * N + col] = f2bf(tanhf(acc[t][v] + bv));
    }
  }
}

// ---------------- GEMM2 + softmax + prior-weighted head mix ----------------
// Lat: [65536][1024] bf16 (rows = token*8+head); Wt: output_w^T bf16 [2048][1024]
// One block per 16 rows (= 2 tokens x 8 heads). 8 waves; wave owns 256 of 2048 cols
// (16 C tiles). V staged in 4 LDS quarters of 512 cols per K-step:
//   B quarter  -> Tensor Data Mover (one descriptor, hardware-padded LDS rows)
//   A tile     -> async global->LDS loads (ASYNCcnt)
__global__ __launch_bounds__(256) void k_mos_softmax(
    const unsigned short* __restrict__ Lat, const unsigned short* __restrict__ Wt,
    const float* __restrict__ obias, const float* __restrict__ prior,
    float* __restrict__ Out) {
  const int K = 1024, V = 2048;
  __shared__ unsigned short sA[16][40];
  __shared__ unsigned short sB[512][40];
  __shared__ float red[8][16];
  const int tid = threadIdx.x;
  const int lane = tid & 31, wave = tid >> 5;
  const int khalf = lane >> 4;
  const size_t rowbase = (size_t)blockIdx.x * 16;
  const unsigned int sB_off = (unsigned int)(unsigned long long)&sB[0][0];
  v8f acc[16] = {};

  for (int k0 = 0; k0 < K; k0 += 32) {
    for (int h = 0; h < 4; ++h) {
      __syncthreads();
      if (h == 0) {  // stage A 16x32 asynchronously (2 elems / 4B per thread)
        int e = tid * 2, m = e >> 5, kk = e & 31;
        async_ld_b32(&sA[m][kk], &Lat[(rowbase + m) * (size_t)K + k0 + kk]);
      }
      if (tid == 0) {  // one TDM descriptor moves the whole 512x32 B quarter
        tdm_load_2d_bf16(sB_off, &Wt[(size_t)(h * 512) * K + k0],
                         /*tensor_d0=*/K, /*tensor_d1=*/V,
                         /*tile_d0=*/32, /*tile_d1=*/512, /*stride_d0=*/K);
      }
      if (wave == 0) __builtin_amdgcn_s_wait_tensorcnt(0);
      wait_asynccnt0();
      __builtin_prefetch((const void*)&Wt[(size_t)(h * 512) * K + k0 + 32], 0, 1);
      __syncthreads();
      v16bf af = ldfragA(&sA[lane & 15][0], khalf);
      v16bf bfm[4];
#pragma unroll
      for (int j = 0; j < 4; ++j)
        bfm[j] = ldfragB(&sB[wave * 64 + j * 16 + (lane & 15)][0], khalf);
#pragma unroll
      for (int j = 0; j < 4; ++j)
        acc[h * 4 + j] = __builtin_amdgcn_wmma_f32_16x16x32_bf16(
            false, af, false, bfm[j], (short)0, acc[h * 4 + j], false, false);
    }
  }

  // add output bias
#pragma unroll
  for (int t = 0; t < 16; ++t) {
    int col = (t >> 2) * 512 + wave * 64 + (t & 3) * 16 + (lane & 15);
    float bo = obias[col];
#pragma unroll
    for (int v = 0; v < 8; ++v) acc[t][v] += bo;
  }

  // per-row max: local over 16 tiles, then 16-lane shuffle tree, then cross-wave via LDS
  float mx8[8];
#pragma unroll
  for (int v = 0; v < 8; ++v) {
    float m = -3.0e38f;
#pragma unroll
    for (int t = 0; t < 16; ++t) m = fmaxf(m, acc[t][v]);
#pragma unroll
    for (int off = 8; off >= 1; off >>= 1) m = fmaxf(m, __shfl_xor(m, off, 32));
    mx8[v] = m;
  }
  if ((lane & 15) == 0) {
#pragma unroll
    for (int v = 0; v < 8; ++v) red[wave][khalf * 8 + v] = mx8[v];
  }
  __syncthreads();
#pragma unroll
  for (int v = 0; v < 8; ++v) {
    float m = red[0][khalf * 8 + v];
#pragma unroll
    for (int w = 1; w < 8; ++w) m = fmaxf(m, red[w][khalf * 8 + v]);
    mx8[v] = m;
  }
  __syncthreads();

  // exp in place + per-row sum (same reduction pattern)
  float sm8[8];
#pragma unroll
  for (int v = 0; v < 8; ++v) {
    float s = 0.f;
#pragma unroll
    for (int t = 0; t < 16; ++t) {
      float e = __expf(acc[t][v] - mx8[v]);
      acc[t][v] = e;
      s += e;
    }
#pragma unroll
    for (int off = 8; off >= 1; off >>= 1) s += __shfl_xor(s, off, 32);
    sm8[v] = s;
  }
  if ((lane & 15) == 0) {
#pragma unroll
    for (int v = 0; v < 8; ++v) red[wave][khalf * 8 + v] = sm8[v];
  }
  __syncthreads();

  const size_t tok = 2 * (size_t)blockIdx.x + khalf;  // lanes 0-15: token0, 16-31: token1
  float wgt[8];
#pragma unroll
  for (int v = 0; v < 8; ++v) {
    float s = 0.f;
#pragma unroll
    for (int w = 0; w < 8; ++w) s += red[w][khalf * 8 + v];
    wgt[v] = prior[tok * 8 + v] / s;  // head v weight / softmax denom
  }
  // per-lane: its 8 C registers are the 8 heads of one token at one column
#pragma unroll
  for (int t = 0; t < 16; ++t) {
    int col = (t >> 2) * 512 + wave * 64 + (t & 3) * 16 + (lane & 15);
    float o = 0.f;
#pragma unroll
    for (int v = 0; v < 8; ++v) o += wgt[v] * acc[t][v];
    Out[tok * (size_t)V + col] = o;
  }
}

// ---------------- launcher ----------------
extern "C" void kernel_launch(void* const* d_in, const int* in_sizes, int n_in,
                              void* d_out, int out_size, void* d_ws, size_t ws_size,
                              hipStream_t stream) {
  (void)in_sizes; (void)n_in; (void)out_size; (void)ws_size;
  const float* hidden   = (const float*)d_in[0];
  const float* prior_w  = (const float*)d_in[1];
  const float* prior_b  = (const float*)d_in[2];
  const float* latent_w = (const float*)d_in[3];
  const float* latent_b = (const float*)d_in[4];
  const float* output_w = (const float*)d_in[5];
  const float* output_b = (const float*)d_in[6];
  float* out = (float*)d_out;

  const int TOK = 8192, H = 1024, NH = 8, V = 2048, NHH = 8192;

  char* ws = (char*)d_ws;
  size_t off = 0;
  auto take = [&](size_t bytes) {
    char* p = ws + off;
    off += (bytes + 255) & ~(size_t)255;
    return p;
  };
  unsigned short* hB  = (unsigned short*)take((size_t)TOK * H * 2);    // hidden bf16
  unsigned short* lwT = (unsigned short*)take((size_t)NHH * H * 2);    // latent_w^T bf16
  unsigned short* owT = (unsigned short*)take((size_t)V * H * 2);      // output_w^T bf16
  float*          pri = (float*)take((size_t)TOK * NH * 4);            // prior
  unsigned short* lat = (unsigned short*)take((size_t)TOK * NHH * 2);  // latent bf16

  k_cvt_bf16<<<(TOK * H) / 256, 256, 0, stream>>>(hidden, hB, (size_t)TOK * H);
  k_transpose_cvt<<<(H * NHH) / 256, 256, 0, stream>>>(latent_w, lwT, H, NHH);
  k_transpose_cvt<<<(H * V) / 256, 256, 0, stream>>>(output_w, owT, H, V);
  k_prior<<<TOK / 8, 256, 0, stream>>>(hidden, prior_w, prior_b, pri);
  k_latent_gemm_tanh<<<dim3(NHH / 256, TOK / 64), 256, 0, stream>>>(hB, lwT, latent_b, lat);
  k_mos_softmax<<<(TOK * NH) / 16, 256, 0, stream>>>(lat, owT, output_b, pri, out);
}
